// CNNLSTMActor_24945170055593
// MI455X (gfx1250) — matrix operand order repaired and checked by reading, not compiled, not run on hardware
//
#include <hip/hip_runtime.h>
#include <hip/hip_bf16.h>
#include <cstdint>

typedef __attribute__((ext_vector_type(16))) _Float16 v16h;
typedef __attribute__((ext_vector_type(8)))  float    v8f;
typedef unsigned int u32;

union Frag { u32 u[8]; v16h v; };
union H2   { _Float16 h[2]; u32 u; };

__device__ __forceinline__ v8f wmma32(const Frag& a, const Frag& b, v8f acc) {
  // D = A(16x32 f16) x B(32x16 f16) + C(f32)
  return __builtin_amdgcn_wmma_f32_16x16x32_f16(false, a.v, false, b.v,
                                                (short)0, acc, false, false);
}

__device__ __forceinline__ float sigmoidf_(float x) { return 1.f / (1.f + __expf(-x)); }

// ---------------- elementwise prep ----------------
__global__ void k_scale_f32_to_f16(const float* __restrict__ in, _Float16* __restrict__ out,
                                   long n, float scale) {
  long i = (long)blockIdx.x * blockDim.x + threadIdx.x;
  if (i < n) out[i] = (_Float16)(in[i] * scale);
}

// Pack conv weights (OIDHW f32) into per-lane swizzled B fragments:
// layout dword index = ((c*nTiles + nt)*32 + lane)*8 + j
// k0 = c*32 + (lane&16?8:0) + 2(j&3) + 16(j>>2)  (k = tap*Cin + cin), n = nt*16 + (lane&15)
__global__ void k_pack_conv_wsz(const float* __restrict__ w, _Float16* __restrict__ wsz,
                                int Cin, int Cout, int Ktot, int nChunks) {
  int idx = blockIdx.x * blockDim.x + threadIdx.x;
  int nTiles = Cout >> 4;
  if (idx >= nChunks * nTiles * 256) return;
  int j = idx & 7, lane = (idx >> 3) & 31;
  int rest = idx >> 8;
  int nt = rest % nTiles, c = rest / nTiles;
  int k0 = c * 32 + ((lane & 16) ? 8 : 0) + 2 * (j & 3) + 16 * (j >> 2);
  int n  = nt * 16 + (lane & 15);
  float v0 = 0.f, v1 = 0.f;
  if (k0 < Ktot) {
    int tap = k0 / Cin;
    int cin = k0 - tap * Cin;           // even; cin+1 same tap (Cin even)
    v0 = w[((long)n * Cin + cin) * 18 + tap];
    v1 = w[((long)n * Cin + cin + 1) * 18 + tap];
  }
  wsz[2 * idx]     = (_Float16)v0;
  wsz[2 * idx + 1] = (_Float16)v1;
}

// Pack GEMM weight W (N x K row-major, B = W^T) into the same swizzled fragment layout.
__global__ void k_pack_gemm_wsz(const float* __restrict__ w, _Float16* __restrict__ wsz,
                                int N, int K) {
  int idx = blockIdx.x * blockDim.x + threadIdx.x;
  int nTiles = N >> 4;
  if (idx >= (K >> 5) * nTiles * 256) return;
  int j = idx & 7, lane = (idx >> 3) & 31;
  int rest = idx >> 8;
  int nt = rest % nTiles, c = rest / nTiles;
  int k0 = c * 32 + ((lane & 16) ? 8 : 0) + 2 * (j & 3) + 16 * (j >> 2);
  int n  = nt * 16 + (lane & 15);
  wsz[2 * idx]     = (_Float16)w[(long)n * K + k0];
  wsz[2 * idx + 1] = (_Float16)w[(long)n * K + k0 + 1];
}

// ------------- generic implicit-GEMM conv (per-k decode) — used only for conv1a (Ci=4) -------------
template <int NT>
__global__ void k_conv3d_wmma_gen(const _Float16* __restrict__ in,
                                  const u32* __restrict__ wsz,
                                  const float* __restrict__ bias,
                                  _Float16* __restrict__ out,
                                  int Ti, int Hi, int Wi, int Ci, int log2Ci,
                                  int To, int Ho, int Wo,
                                  int sh, int sw, int Ktot, int nChunks, float outScale) {
  const int lane  = threadIdx.x;
  const int tileM = blockIdx.x;
  int m = tileM * 16 + (lane & 15);
  int xo = m % Wo; int t1 = m / Wo;
  int yo = t1 % Ho; int t2 = t1 / Ho;
  int to = t2 % To; int nb = t2 / To;
  const int hiK = (lane & 16) ? 8 : 0;
  v8f acc[NT];
#pragma unroll
  for (int nt = 0; nt < NT; ++nt) acc[nt] = {};
#pragma unroll 1
  for (int c = 0; c < nChunks; ++c) {
    Frag a;
    int kb = c * 32 + hiK;
#pragma unroll
    for (int j = 0; j < 8; ++j) {
      int k = kb + (j & 3) * 2 + (j >> 2) * 16;
      u32 val = 0u;
      if (k < Ktot) {
        int tap = k >> log2Ci;
        int cin = k & (Ci - 1);
        int kd = tap / 9; int r = tap - kd * 9;
        int kh = r / 3;   int kw = r - kh * 3;
        int tin = to + kd;
        int yin = yo * sh + kh - 1;
        int xin = xo * sw + kw - 1;
        if (yin >= 0 && yin < Hi && xin >= 0 && xin < Wi) {
          long offn = ((((long)nb * Ti + tin) * Hi + yin) * Wi + xin) * Ci + cin;
          val = *(const u32*)(in + offn);
        }
      }
      a.u[j] = val;
    }
    const u32* bp = wsz + ((long)c * NT * 32 + lane) * 8;
#pragma unroll
    for (int nt = 0; nt < NT; ++nt) {
      Frag b;
#pragma unroll
      for (int j = 0; j < 8; ++j) b.u[j] = bp[nt * 256 + j];
      acc[nt] = wmma32(a, b, acc[nt]);
    }
  }
  int mBase = tileM * 16 + ((lane & 16) ? 8 : 0);
#pragma unroll
  for (int v = 0; v < 8; ++v) {
    int mm = mBase + v;
    int x2 = mm % Wo; int u1 = mm / Wo;
    int y2 = u1 % Ho; int u2 = u1 / Ho;
    int t3 = u2 % To; int n3 = u2 / To;
    long obase = ((((long)n3 * To + t3) * Ho + y2) * Wo + x2) * (NT * 16);
#pragma unroll
    for (int nt = 0; nt < NT; ++nt) {
      int nCol = nt * 16 + (lane & 15);
      float val = fmaxf(acc[nt][v] + bias[nCol], 0.f) * outScale;
      out[obase + nCol] = (_Float16)val;
    }
  }
}

// ------------- fast implicit-GEMM conv: Ci multiple of 32, tap decode hoisted -------------
// NT = Co/16 accumulator tiles, SUBS = Ci/32 K-chunks per tap.
template <int NT, int SUBS>
__global__ void k_conv3d_wmma_big(const _Float16* __restrict__ in,
                                  const u32* __restrict__ wsz,
                                  const float* __restrict__ bias,
                                  _Float16* __restrict__ out,
                                  int Ti, int Hi, int Wi,
                                  int To, int Ho, int Wo,
                                  int sh, int sw, float outScale) {
  const int Ci   = SUBS * 32;
  const int lane = threadIdx.x;
  const int tileM = blockIdx.x;
  int m = tileM * 16 + (lane & 15);
  int xo = m % Wo; int t1 = m / Wo;
  int yo = t1 % Ho; int t2 = t1 / Ho;
  int to = t2 % To; int nb = t2 / To;
  const int cb0 = (lane & 16) ? 4 : 0;     // A dword offset within chunk
  v8f acc[NT];
#pragma unroll
  for (int nt = 0; nt < NT; ++nt) acc[nt] = {};

  int tap = 0, kd = 0, kh = 0, kw = 0;
#pragma unroll 1
  for (tap = 0; tap < 18; ++tap) {
    int tin = to + kd;
    int yin = yo * sh + kh - 1;
    int xin = xo * sw + kw - 1;
    bool ok = (yin >= 0) & (yin < Hi) & (xin >= 0) & (xin < Wi);
    long offn = ((((long)nb * Ti + tin) * Hi + (ok ? yin : 0)) * Wi + (ok ? xin : 0)) * Ci;
    const u32* ap = (const u32*)(in + offn);
#pragma unroll
    for (int sub = 0; sub < SUBS; ++sub) {
      Frag a;
      int cbase = sub * 16 + cb0;
      if (ok) {
#pragma unroll
        for (int j = 0; j < 8; ++j) a.u[j] = ap[cbase + (j & 3) + (j >> 2) * 8];
      } else {
#pragma unroll
        for (int j = 0; j < 8; ++j) a.u[j] = 0u;
      }
      int c = tap * SUBS + sub;
      const u32* bp = wsz + ((long)c * NT * 32 + lane) * 8;
#pragma unroll
      for (int nt = 0; nt < NT; ++nt) {
        Frag b;
#pragma unroll
        for (int j = 0; j < 8; ++j) b.u[j] = bp[nt * 256 + j];
        acc[nt] = wmma32(a, b, acc[nt]);
      }
    }
    if (++kw == 3) { kw = 0; if (++kh == 3) { kh = 0; ++kd; } }
  }

  int mBase = tileM * 16 + ((lane & 16) ? 8 : 0);
#pragma unroll
  for (int v = 0; v < 8; ++v) {
    int mm = mBase + v;
    int x2 = mm % Wo; int u1 = mm / Wo;
    int y2 = u1 % Ho; int u2 = u1 / Ho;
    int t3 = u2 % To; int n3 = u2 / To;
    long obase = ((((long)n3 * To + t3) * Ho + y2) * Wo + x2) * (NT * 16);
#pragma unroll
    for (int nt = 0; nt < NT; ++nt) {
      int nCol = nt * 16 + (lane & 15);
      float val = fmaxf(acc[nt][v] + bias[nCol], 0.f) * outScale;
      out[obase + nCol] = (_Float16)val;
    }
  }
}

// ---------------- faithful-reshape gather: conv3b (n,t,y,x,c) -> X[5][256][2048] ----------------
__global__ void k_gather_x(const _Float16* __restrict__ act, _Float16* __restrict__ X) {
  long i = (long)blockIdx.x * blockDim.x + threadIdx.x;
  if (i >= 5L * 256 * 2048) return;
  long bsrc = i / 10240;                 // flat (256,10240) row
  int k = (int)(i - bsrc * 10240);       // k = c*80 + t*8 + y*4 + x (C-major CDHW)
  int c = k / 80; int r = k - c * 80;
  int t = r >> 3; int r2 = r & 7;
  int y = r2 >> 2; int x = r2 & 3;
  X[i] = act[((((bsrc * 10 + t) * 2 + y) * 4 + x)) * 128 + c];
}

// ---------------- LSTM gates GEMM: gates = X*Wih^T + (h*mask)*Whh^T, 4 N-tiles per wave ----------------
__global__ void k_lstm_gates(const _Float16* __restrict__ X,    // 256 x 2048 f16
                             const u32* __restrict__ wih,       // swizzled [64][128][32][8]
                             const u32* __restrict__ whh,       // swizzled [16][128][32][8]
                             const float* __restrict__ hstate,  // 256 x 512 f32
                             const float* __restrict__ done_l,  // 256
                             float* __restrict__ gates) {       // 256 x 2048 f32
  const int lane  = threadIdx.x;
  const int tileM = blockIdx.x, tileN = blockIdx.y;   // grid (16, 32)
  const int row   = tileM * 16 + (lane & 15);
  const int cb0   = (lane & 16) ? 4 : 0;
  const u32* xrow = (const u32*)(X + (long)row * 2048);
  v8f acc[4];
#pragma unroll
  for (int nt = 0; nt < 4; ++nt) acc[nt] = {};
#pragma unroll 1
  for (int c = 0; c < 64; ++c) {          // K = 2048
    Frag a;
    int cbase = c * 16 + cb0;
#pragma unroll
    for (int j = 0; j < 8; ++j) a.u[j] = xrow[cbase + (j & 3) + (j >> 2) * 8];
    const u32* bp = wih + (((long)c * 128 + tileN * 4) * 32 + lane) * 8;
    if (c + 1 < 64) __builtin_prefetch(bp + 128 * 256, 0, 0);
#pragma unroll
    for (int nt = 0; nt < 4; ++nt) {
      Frag b;
#pragma unroll
      for (int j = 0; j < 8; ++j) b.u[j] = bp[nt * 256 + j];
      acc[nt] = wmma32(a, b, acc[nt]);
    }
  }
  float mask = 1.f - done_l[row];
  const float* hrow = hstate + (long)row * 512;
#pragma unroll 1
  for (int c = 0; c < 16; ++c) {          // K = 512
    Frag a;
    int kb = c * 32 + 2 * cb0;
#pragma unroll
    for (int j = 0; j < 8; ++j) {
      int k = kb + (j & 3) * 2 + (j >> 2) * 16;
      H2 p; p.h[0] = (_Float16)(hrow[k] * mask); p.h[1] = (_Float16)(hrow[k + 1] * mask);
      a.u[j] = p.u;
    }
    const u32* bp = whh + (((long)c * 128 + tileN * 4) * 32 + lane) * 8;
#pragma unroll
    for (int nt = 0; nt < 4; ++nt) {
      Frag b;
#pragma unroll
      for (int j = 0; j < 8; ++j) b.u[j] = bp[nt * 256 + j];
      acc[nt] = wmma32(a, b, acc[nt]);
    }
  }
  int mBase = tileM * 16 + ((lane & 16) ? 8 : 0);
#pragma unroll
  for (int v = 0; v < 8; ++v) {
#pragma unroll
    for (int nt = 0; nt < 4; ++nt)
      gates[(long)(mBase + v) * 2048 + tileN * 64 + nt * 16 + (lane & 15)] = acc[nt][v];
  }
}

// ---------------- LSTM cell pointwise ----------------
__global__ void k_lstm_cell(const float* __restrict__ gates,
                            const float* __restrict__ b_ih, const float* __restrict__ b_hh,
                            const float* __restrict__ done_l,
                            float* __restrict__ hstate, float* __restrict__ cstate,
                            _Float16* __restrict__ hsh_l) {
  int idx = blockIdx.x * blockDim.x + threadIdx.x;
  if (idx >= 256 * 512) return;
  int b = idx >> 9, u = idx & 511;
  const float* g = gates + (long)b * 2048;
  float gi = g[u]        + b_ih[u]        + b_hh[u];
  float gf = g[u + 512]  + b_ih[u + 512]  + b_hh[u + 512];
  float gg = g[u + 1024] + b_ih[u + 1024] + b_hh[u + 1024];
  float go = g[u + 1536] + b_ih[u + 1536] + b_hh[u + 1536];
  float mask  = 1.f - done_l[b];
  float cprev = cstate[idx] * mask;
  float cn = sigmoidf_(gf) * cprev + sigmoidf_(gi) * tanhf(gg);
  float hn = sigmoidf_(go) * tanhf(cn);
  cstate[idx] = cn;
  hstate[idx] = hn;
  hsh_l[idx]  = (_Float16)hn;
}

// ---------------- fc1: Z = relu(hs * fc1^T + b), WMMA, 4 N-tiles per wave ----------------
__global__ void k_fc1_wmma(const _Float16* __restrict__ A,   // 1280 x 512 f16
                           const u32* __restrict__ wp,       // swizzled [16][32][32][8]
                           const float* __restrict__ bias,
                           _Float16* __restrict__ Z) {       // 1280 x 512 f16
  const int lane  = threadIdx.x;
  const int tileM = blockIdx.x, tileN = blockIdx.y;  // grid (80, 8)
  const int row   = tileM * 16 + (lane & 15);
  const int cb0   = (lane & 16) ? 4 : 0;
  const u32* arow = (const u32*)(A + (long)row * 512);
  v8f acc[4];
#pragma unroll
  for (int nt = 0; nt < 4; ++nt) acc[nt] = {};
#pragma unroll 1
  for (int c = 0; c < 16; ++c) {
    Frag a;
    int cbase = c * 16 + cb0;
#pragma unroll
    for (int j = 0; j < 8; ++j) a.u[j] = arow[cbase + (j & 3) + (j >> 2) * 8];
    const u32* bp = wp + (((long)c * 32 + tileN * 4) * 32 + lane) * 8;
#pragma unroll
    for (int nt = 0; nt < 4; ++nt) {
      Frag b;
#pragma unroll
      for (int j = 0; j < 8; ++j) b.u[j] = bp[nt * 256 + j];
      acc[nt] = wmma32(a, b, acc[nt]);
    }
  }
  int mBase = tileM * 16 + ((lane & 16) ? 8 : 0);
#pragma unroll
  for (int v = 0; v < 8; ++v) {
#pragma unroll
    for (int nt = 0; nt < 4; ++nt) {
      int nCol = tileN * 64 + nt * 16 + (lane & 15);
      Z[(long)(mBase + v) * 512 + nCol] = (_Float16)fmaxf(acc[nt][v] + bias[nCol], 0.f);
    }
  }
}

// ---------------- fc2 (N=5): plain VALU dot ----------------
__global__ void k_fc2(const _Float16* __restrict__ Z, const float* __restrict__ W,
                      const float* __restrict__ b, float* __restrict__ outv) {
  int idx = blockIdx.x * blockDim.x + threadIdx.x;
  if (idx >= 1280 * 5) return;
  int mrow = idx / 5, j = idx - mrow * 5;
  const _Float16* z = Z + (long)mrow * 512;
  const float* w = W + (long)j * 512;
  float acc = 0.f;
  for (int k = 0; k < 512; ++k) acc += (float)z[k] * w[k];
  outv[idx] = acc + b[j];
}

extern "C" void kernel_launch(void* const* d_in, const int* in_sizes, int n_in,
                              void* d_out, int out_size, void* d_ws, size_t ws_size,
                              hipStream_t stream) {
  (void)in_sizes; (void)n_in; (void)out_size; (void)ws_size;
  const float* x    = (const float*)d_in[0];
  const float* done = (const float*)d_in[1];
  const float* h0   = (const float*)d_in[2];
  const float* c0   = (const float*)d_in[3];
  const float* w1a = (const float*)d_in[4];  const float* b1a = (const float*)d_in[5];
  const float* w1b = (const float*)d_in[6];  const float* b1b = (const float*)d_in[7];
  const float* w2a = (const float*)d_in[8];  const float* b2a = (const float*)d_in[9];
  const float* w2b = (const float*)d_in[10]; const float* b2b = (const float*)d_in[11];
  const float* w3a = (const float*)d_in[12]; const float* b3a = (const float*)d_in[13];
  const float* w3b = (const float*)d_in[14]; const float* b3b = (const float*)d_in[15];
  const float* Wih = (const float*)d_in[16]; const float* Whh = (const float*)d_in[17];
  const float* bih = (const float*)d_in[18]; const float* bhh = (const float*)d_in[19];
  const float* fc1w = (const float*)d_in[20]; const float* fc1b = (const float*)d_in[21];
  const float* fc2w = (const float*)d_in[22]; const float* fc2b = (const float*)d_in[23];
  float* out = (float*)d_out;

  // workspace carve-up (~174 MB), 256B aligned
  char* ws = (char*)d_ws;
  size_t off = 0;
  auto alloc = [&](size_t bytes) -> char* {
    char* p = ws + off; off = (off + bytes + 255) & ~(size_t)255; return p;
  };
  _Float16* regA = (_Float16*)alloc(62914560ull * 2);  // act1a/2a/3a ping
  _Float16* regB = (_Float16*)alloc(14680064ull * 2);  // x16/act1b/2b/3b pong
  _Float16* wp1a = (_Float16*)alloc((size_t)3  * 2 * 256 * 4);
  _Float16* wp1b = (_Float16*)alloc((size_t)18 * 2 * 256 * 4);
  _Float16* wp2a = (_Float16*)alloc((size_t)18 * 4 * 256 * 4);
  _Float16* wp2b = (_Float16*)alloc((size_t)36 * 4 * 256 * 4);
  _Float16* wp3a = (_Float16*)alloc((size_t)36 * 8 * 256 * 4);
  _Float16* wp3b = (_Float16*)alloc((size_t)72 * 8 * 256 * 4);
  _Float16* wihp = (_Float16*)alloc(64ull * 128 * 256 * 4);
  _Float16* whhp = (_Float16*)alloc(16ull * 128 * 256 * 4);
  _Float16* fc1p = (_Float16*)alloc(16ull * 32 * 256 * 4);
  _Float16* Xl   = (_Float16*)alloc(2621440ull * 2);
  _Float16* hsh  = (_Float16*)alloc(655360ull * 2);
  float* gates   = (float*)alloc(524288ull * 4);       // reused as fc1 Z (f16) later
  float* hstate  = (float*)alloc(131072ull * 4);
  float* cstate  = (float*)alloc(131072ull * 4);

  // x (B,T,H,W,C) f32 -> f16 * (1/255)  (channels-last already; no transpose needed)
  {
    long n = 256L * 16 * 16 * 32 * 4;
    k_scale_f32_to_f16<<<(int)((n + 255) / 256), 256, 0, stream>>>(x, regB, n, 1.f / 255.f);
  }
  // pack weights into per-lane swizzled fragment layouts
  k_pack_conv_wsz<<<(3 * 2 * 256 + 255) / 256, 256, 0, stream>>>(w1a, wp1a, 4, 32, 72, 3);
  k_pack_conv_wsz<<<(18 * 2 * 256 + 255) / 256, 256, 0, stream>>>(w1b, wp1b, 32, 32, 576, 18);
  k_pack_conv_wsz<<<(18 * 4 * 256 + 255) / 256, 256, 0, stream>>>(w2a, wp2a, 32, 64, 576, 18);
  k_pack_conv_wsz<<<(36 * 4 * 256 + 255) / 256, 256, 0, stream>>>(w2b, wp2b, 64, 64, 1152, 36);
  k_pack_conv_wsz<<<(36 * 8 * 256 + 255) / 256, 256, 0, stream>>>(w3a, wp3a, 64, 128, 1152, 36);
  k_pack_conv_wsz<<<(72 * 8 * 256 + 255) / 256, 256, 0, stream>>>(w3b, wp3b, 128, 128, 2304, 72);
  k_pack_gemm_wsz<<<(64 * 128 * 256 + 255) / 256, 256, 0, stream>>>(Wih, wihp, 2048, 2048);
  k_pack_gemm_wsz<<<(16 * 128 * 256 + 255) / 256, 256, 0, stream>>>(Whh, whhp, 2048, 512);
  k_pack_gemm_wsz<<<(16 * 32 * 256 + 255) / 256, 256, 0, stream>>>(fc1w, fc1p, 512, 512);

  // conv chain: one wave computes 16 M-rows x all Cout tiles; /255 fused at block boundaries
  k_conv3d_wmma_gen<2><<<122880, 32, 0, stream>>>(regB, (const u32*)wp1a, b1a, regA,
      16, 16, 32, 4, 2,  15, 16, 32,  1, 1,  72, 3, 1.f);
  k_conv3d_wmma_big<2, 1><<<28672, 32, 0, stream>>>(regA, (const u32*)wp1b, b1b, regB,
      15, 16, 32,  14, 8, 16,  2, 2, 1.f / 255.f);
  k_conv3d_wmma_big<4, 1><<<26624, 32, 0, stream>>>(regB, (const u32*)wp2a, b2a, regA,
      14, 8, 16,  13, 8, 16,  1, 1, 1.f);
  k_conv3d_wmma_big<4, 2><<<6144, 32, 0, stream>>>(regA, (const u32*)wp2b, b2b, regB,
      13, 8, 16,  12, 4, 8,  2, 2, 1.f / 255.f);
  k_conv3d_wmma_big<8, 2><<<5632, 32, 0, stream>>>(regB, (const u32*)wp3a, b3a, regA,
      12, 4, 8,  11, 4, 8,  1, 1, 1.f);
  k_conv3d_wmma_big<8, 4><<<1280, 32, 0, stream>>>(regA, (const u32*)wp3b, b3b, regB,
      11, 4, 8,  10, 2, 4,  2, 2, 1.f);

  // faithful (5,256,2048) reshape gather
  k_gather_x<<<(2621440 + 255) / 256, 256, 0, stream>>>(regB, Xl);

  // LSTM state init (do not mutate inputs)
  hipMemcpyAsync(hstate, h0, 131072 * 4, hipMemcpyDeviceToDevice, stream);
  hipMemcpyAsync(cstate, c0, 131072 * 4, hipMemcpyDeviceToDevice, stream);

  for (int l = 0; l < 5; ++l) {
    k_lstm_gates<<<dim3(16, 32), 32, 0, stream>>>(Xl + (long)l * 524288,
        (const u32*)wihp, (const u32*)whhp, hstate, done + l * 256, gates);
    k_lstm_cell<<<512, 256, 0, stream>>>(gates, bih, bhh, done + l * 256,
        hstate, cstate, hsh + (long)l * 131072);
  }

  // fc1 (WMMA) -> Z f16 (reuse gates buffer), fc2 (VALU, N=5) -> logits
  k_fc1_wmma<<<dim3(80, 8), 32, 0, stream>>>(hsh, (const u32*)fc1p, fc1b, (_Float16*)gates);
  k_fc2<<<25, 256, 0, stream>>>((const _Float16*)gates, fc2w, fc2b, out);

  // outputs: logits(1280x5) | hT(256x512) | cT(256x512)
  hipMemcpyAsync(out + 6400, hstate, 131072 * 4, hipMemcpyDeviceToDevice, stream);
  hipMemcpyAsync(out + 6400 + 131072, cstate, 131072 * 4, hipMemcpyDeviceToDevice, stream);
}
